// EmbeddingMulti_76630806495461
// MI455X (gfx1250) — matrix operand order repaired and checked by reading, compiled, not run on hardware
//
#include <hip/hip_runtime.h>
#include <hip/hip_bf16.h>

// ---------------------------------------------------------------------------
// Problem constants (from reference): BATCH=32768, NUM_GENRES=1000, EMB_DIM=128
// out[b, :] = broadcast( (mask[b,:] @ row_sum) / (count[b] * 128) )
// ---------------------------------------------------------------------------
#define BATCH     32768
#define GENRES    1000
#define KPAD      1024      // GENRES padded to multiple of 32 (WMMA K)
#define KSTEPS    (KPAD / 32)
#define EMB_DIM   128

typedef __attribute__((ext_vector_type(16))) _Float16 v16h;
typedef __attribute__((ext_vector_type(8)))  float    v8f;

// ---------------------------------------------------------------------------
// Kernel 1: row_sum[g] = sum_d emb[g][d]; split into f16 hi/lo and store as
// packed half2 pairs:  pb[0..511] = hi pairs, pb[512..1023] = lo pairs.
// Genres >= 1000 are zero-padded (so padded WMMA K-rows contribute nothing).
// ---------------------------------------------------------------------------
__global__ __launch_bounds__(256) void genre_rowsum_pack_kernel(
    const float* __restrict__ emb, unsigned int* __restrict__ pb) {
  int t = blockIdx.x * 256 + threadIdx.x;   // 0..511, pair index
  if (t >= KPAD / 2) return;
  float s[2];
#pragma unroll
  for (int j = 0; j < 2; ++j) {
    int g = 2 * t + j;
    float acc = 0.0f;
    if (g < GENRES) {
      const float4* p = (const float4*)(emb + (size_t)g * EMB_DIM);
#pragma unroll 8
      for (int i = 0; i < EMB_DIM / 4; ++i) {
        float4 q = p[i];
        acc += (q.x + q.y) + (q.z + q.w);
      }
    }
    s[j] = acc;
  }
  _Float16 h0 = (_Float16)s[0];
  _Float16 l0 = (_Float16)(s[0] - (float)h0);
  _Float16 h1 = (_Float16)s[1];
  _Float16 l1 = (_Float16)(s[1] - (float)h1);
  union { _Float16 h[2]; unsigned int u; } ph, pl;
  ph.h[0] = h0; ph.h[1] = h1;
  pl.h[0] = l0; pl.h[1] = l1;
  pb[t]            = ph.u;   // hi region
  pb[KPAD / 2 + t] = pl.u;   // lo region
}

// Pack two {0,1} ints into half2 bits {0x0000/0x3C00, 0x0000/0x3C00}.
// Inputs are guaranteed in {0,1} (reference setup: randint(0,2)), so
// x * 0x3C00 per 16-bit half == (x != 0 ? 1.0h : 0.0h).
// 2 full-rate VALU ops: v_lshl_or_b32 + v_pk_mul_lo_u16 (VOP3P).
__device__ __forceinline__ unsigned int pack01(int x0, int x1) {
  unsigned int t = (unsigned int)x0 | ((unsigned int)x1 << 16);
  unsigned int c = 0x3C003C00u;
  unsigned int r;
  asm("v_pk_mul_lo_u16 %0, %1, %2" : "=v"(r) : "v"(t), "v"(c));
  return r;
}

__device__ __forceinline__ float lane_bcast(float v, int srcLane) {
  int r = __builtin_amdgcn_ds_bpermute(srcLane << 2,
                                       __builtin_bit_cast(int, v));
  return __builtin_bit_cast(float, r);
}

// ---------------------------------------------------------------------------
// Kernel 2: one wave <-> 16 batch rows. D(16x16,f32) = A(16x32 f16 mask-tile)
//           x B(32x16 f16: col0=rs_hi, col1=rs_lo, col2=1.0, rest 0), looped
//           over K with v_wmma_f32_16x16x32_f16.
// The full per-(kstep,lane) B register image is prebuilt in LDS once per
// block, so the hot loop is branch-free: 4x global_load_b128 +
// 2x ds_load_b128 + 8x (lshl_or + pk_mul) + 1 wmma.
// ---------------------------------------------------------------------------
__global__ __launch_bounds__(256) void embmulti_wmma_kernel(
    const int* __restrict__ inp, const unsigned int* __restrict__ pb,
    float* __restrict__ out) {
  // B fragment image: [kstep][lane][8 dwords] = 32*32*8 dwords = 32 KB
  __shared__ unsigned int lds_bfrag[KSTEPS * 32 * 8];

  const int tid = threadIdx.x;

  // ---- one-time prologue: materialize B fragments (branches OK here) ----
  for (int idx = tid; idx < KSTEPS * 32 * 8; idx += 256) {
    const int vv = idx & 7;
    const int ln = (idx >> 3) & 31;
    const int ks = idx >> 8;
    const int c  = ln & 15;
    const int hh = ln >> 4;
    const int pi = ks * 16 + hh * 4 + vv + (vv >> 2) * 4;  // packed-pair index into pb
    unsigned int val;
    if (c == 0)      val = pb[pi];                // rs_hi pairs
    else if (c == 1) val = pb[KPAD / 2 + pi];     // rs_lo pairs
    else if (c == 2) val = 0x3C003C00u;           // ones column -> count
    else             val = 0u;
    lds_bfrag[idx] = val;
  }
  __syncthreads();

  const int lane    = tid & 31;
  const int wave    = tid >> 5;
  const int col     = lane & 15;        // A-row within tile
  const int hiHalf  = lane >> 4;        // 0: lanes 0-15, 1: lanes 16-31
  const int rowBase = (blockIdx.x * 8 + wave) * 16;
  const int row     = rowBase + col;
  const int* __restrict__ rowPtr = inp + (size_t)row * GENRES;
  const int koffA = hiHalf * 8;         // A: K offsets {0-7,16-23} vs {8-15,24-31}

  const uint4* __restrict__ bfrag =
      (const uint4*)&lds_bfrag[(size_t)lane << 3];

  v8f acc = {};

  auto wmma_step = [&](int ks, int4 m0, int4 m1, int4 m2, int4 m3) {
    union { v16h v; unsigned int u[8]; } A;
    A.u[0] = pack01(m0.x, m0.y);  A.u[1] = pack01(m0.z, m0.w);
    A.u[2] = pack01(m1.x, m1.y);  A.u[3] = pack01(m1.z, m1.w);
    A.u[4] = pack01(m2.x, m2.y);  A.u[5] = pack01(m2.z, m2.w);
    A.u[6] = pack01(m3.x, m3.y);  A.u[7] = pack01(m3.z, m3.w);
    union { v16h v; uint4 q[2]; } B;
    B.q[0] = bfrag[(size_t)ks * 64 + 0];   // ks*256 dwords / 4 = ks*64 uint4
    B.q[1] = bfrag[(size_t)ks * 64 + 1];
    acc = __builtin_amdgcn_wmma_f32_16x16x32_f16(
        /*neg_a=*/false, A.v, /*neg_b=*/false, B.v,
        /*c_mod=*/(short)0, acc, /*reuse_a=*/false, /*reuse_b=*/false);
  };

  // ---- main loop: k0 <= 960 -> max column touched = k0 + 31 <= 991 < 1000,
  //      so all loads are unconditional ----
#pragma unroll 2
  for (int k0 = 0; k0 < 992; k0 += 32) {
    const int g1 = k0 + koffA;
    int4 m0 = *(const int4*)(rowPtr + g1);
    int4 m1 = *(const int4*)(rowPtr + g1 + 4);
    int4 m2 = *(const int4*)(rowPtr + g1 + 16);
    int4 m3 = *(const int4*)(rowPtr + g1 + 20);
    wmma_step(k0 >> 5, m0, m1, m2, m3);
  }

  // ---- tail: k0 = 992. g1 = 992 (lanes<16, valid: cols 992..999) or 1000
  //      (lanes>=16, out of range). g1+16 groups always out of range. ----
  {
    const int g1 = 992 + koffA;
    int4 z = {0, 0, 0, 0};
    int4 m0 = z, m1 = z;
    if (g1 < GENRES) {
      m0 = *(const int4*)(rowPtr + g1);
      m1 = *(const int4*)(rowPtr + g1 + 4);
    }
    wmma_step(31, m0, m1, z, z);
  }

  // ---- Epilogue: D[m][0]=total_hi, D[m][1]=total_lo, D[m][2]=count ----
  // acc[r]: lanes 0-15 -> row r, lanes 16-31 -> row r+8 (cols = lane&15)
  const float inv_d = 1.0f / (float)EMB_DIM;
#pragma unroll
  for (int r = 0; r < 8; ++r) {
    float a   = acc[r];
    float th0 = lane_bcast(a, 0),  tl0 = lane_bcast(a, 1),  cn0 = lane_bcast(a, 2);
    float th1 = lane_bcast(a, 16), tl1 = lane_bcast(a, 17), cn1 = lane_bcast(a, 18);
    float v0 = ((th0 + tl0) / cn0) * inv_d;   // row rowBase + r
    float v1 = ((th1 + tl1) / cn1) * inv_d;   // row rowBase + r + 8
    float4 o0 = {v0, v0, v0, v0};
    float4 o1 = {v1, v1, v1, v1};
    ((float4*)(out + (size_t)(rowBase + r)     * EMB_DIM))[lane] = o0;
    ((float4*)(out + (size_t)(rowBase + r + 8) * EMB_DIM))[lane] = o1;
  }
}

extern "C" void kernel_launch(void* const* d_in, const int* in_sizes, int n_in,
                              void* d_out, int out_size, void* d_ws, size_t ws_size,
                              hipStream_t stream) {
  const int*   inp = (const int*)d_in[0];    // [32768, 1000] int32 multi-hot
  const float* emb = (const float*)d_in[1];  // [1000, 128] float32
  float*       out = (float*)d_out;          // [32768, 128] float32
  unsigned int* pb = (unsigned int*)d_ws;    // 1024 dwords: hi/lo packed pairs

  genre_rowsum_pack_kernel<<<2, 256, 0, stream>>>(emb, pb);
  // 256 blocks * 8 waves * 16 rows = 32768 rows
  embmulti_wmma_kernel<<<BATCH / 128, 256, 0, stream>>>(inp, pb, out);
}
